// Encoder_4166118277412
// MI455X (gfx1250) — matrix operand order repaired and checked by reading, not compiled, not run on hardware
//
#include <hip/hip_runtime.h>
#include <math.h>

#define N_NODES 50000
#define E_EDGES 1600000
#define DIM     256
#define DIM2    512
#define KEXP    4
#define LLAYERS 2

typedef __attribute__((ext_vector_type(16))) __bf16       v16bf;
typedef __attribute__((ext_vector_type(8)))  float        v8f;
typedef __attribute__((ext_vector_type(4)))  unsigned int u32x4;

union Frag {
    u32x4 u[2];
    v16bf v;
};
union PackPair {
    __bf16 h[2];
    unsigned int u;
};

// ---------------------------------------------------------------------------
// Router: logits = z @ envW[l] + envb[l]; softmax over K=4. One wave per node.
// ---------------------------------------------------------------------------
__global__ void __launch_bounds__(256)
router_kernel(const float* __restrict__ z, const float* __restrict__ envW,
              const float* __restrict__ envb, float* __restrict__ wts) {
    int wid  = threadIdx.x >> 5;
    int lane = threadIdx.x & 31;
    int node = blockIdx.x * 8 + wid;
    if (node >= N_NODES) return;

    const float* zr = z + (size_t)node * DIM;
    float a0 = 0.f, a1 = 0.f, a2 = 0.f, a3 = 0.f;
#pragma unroll
    for (int i = 0; i < 8; ++i) {
        int d = lane + i * 32;
        float zv = zr[d];
        const float* ew = envW + d * KEXP;
        a0 += zv * ew[0]; a1 += zv * ew[1]; a2 += zv * ew[2]; a3 += zv * ew[3];
    }
#pragma unroll
    for (int off = 16; off > 0; off >>= 1) {
        a0 += __shfl_down(a0, off, 32);
        a1 += __shfl_down(a1, off, 32);
        a2 += __shfl_down(a2, off, 32);
        a3 += __shfl_down(a3, off, 32);
    }
    if (lane == 0) {
        float l0 = a0 + envb[0], l1 = a1 + envb[1];
        float l2 = a2 + envb[2], l3 = a3 + envb[3];
        float mx = fmaxf(fmaxf(l0, l1), fmaxf(l2, l3));
        float e0 = __expf(l0 - mx), e1 = __expf(l1 - mx);
        float e2 = __expf(l2 - mx), e3 = __expf(l3 - mx);
        float inv = 1.0f / (e0 + e1 + e2 + e3);
        float* wp = wts + (size_t)node * KEXP;
        wp[0] = e0 * inv; wp[1] = e1 * inv; wp[2] = e2 * inv; wp[3] = e3 * inv;
    }
}

// ---------------------------------------------------------------------------
// CSR build step 1: degree histogram (int atomics, 1 per edge).
// ---------------------------------------------------------------------------
__global__ void __launch_bounds__(256)
deg_kernel(const long long* __restrict__ ei, int* __restrict__ deg) {
    int e = blockIdx.x * 256 + threadIdx.x;
    if (e >= E_EDGES) return;
    int row = (int)ei[e];
    atomicAdd(&deg[row], 1);
}

// ---------------------------------------------------------------------------
// CSR build step 2: exclusive prefix sum over deg[N] -> off[N+1]. Single block.
// ---------------------------------------------------------------------------
__global__ void __launch_bounds__(1024)
scan_kernel(const int* __restrict__ deg, int* __restrict__ off) {
    __shared__ int lds[1024];
    __shared__ int carry_s;
    int tid = threadIdx.x;
    if (tid == 0) carry_s = 0;
    __syncthreads();
    for (int base = 0; base < N_NODES; base += 1024) {
        int i = base + tid;
        int v = (i < N_NODES) ? deg[i] : 0;
        lds[tid] = v;
        __syncthreads();
        for (int s = 1; s < 1024; s <<= 1) {
            int t = (tid >= s) ? lds[tid - s] : 0;
            __syncthreads();
            lds[tid] += t;
            __syncthreads();
        }
        int incl  = lds[tid];
        int carry = carry_s;
        if (i < N_NODES) off[i] = carry + incl - v;
        __syncthreads();
        if (tid == 1023) carry_s = carry + lds[1023];
        __syncthreads();
    }
    if (tid == 0) off[N_NODES] = carry_s;
}

// ---------------------------------------------------------------------------
// CSR build step 3: fill column lists (returning int atomic per edge).
// ---------------------------------------------------------------------------
__global__ void __launch_bounds__(256)
fill_kernel(const long long* __restrict__ ei, const int* __restrict__ off,
            int* __restrict__ cursor, int* __restrict__ csr_col) {
    int e = blockIdx.x * 256 + threadIdx.x;
    if (e >= E_EDGES) return;
    int row = (int)ei[e];
    int col = (int)ei[(long long)E_EDGES + e];
    int pos = atomicAdd(&cursor[row], 1);
    csr_col[off[row] + pos] = col;
}

// ---------------------------------------------------------------------------
// Fused gather-mean + bf16 pack: one wave per node. No float atomics.
// combA[n] = bf16(concat(mean_{c in adj(n)} z[c], z[n]))  as [N,512] bf16.
// ---------------------------------------------------------------------------
__global__ void __launch_bounds__(256)
agg_combine_kernel(const float* __restrict__ z, const int* __restrict__ off,
                   const int* __restrict__ csr_col, unsigned int* __restrict__ combA) {
    int wid  = threadIdx.x >> 5;
    int lane = threadIdx.x & 31;
    int n = blockIdx.x * 8 + wid;
    if (n >= N_NODES) return;

    int s = off[n], e = off[n + 1];
    float ax[4] = {0.f, 0.f, 0.f, 0.f};
    float ay[4] = {0.f, 0.f, 0.f, 0.f};

    int j = s;
    for (; j + 1 < e; j += 2) {           // 2-edge unroll for latency hiding
        int c0 = csr_col[j];
        int c1 = csr_col[j + 1];
        const float2* r0 = (const float2*)(z + (size_t)c0 * DIM);
        const float2* r1 = (const float2*)(z + (size_t)c1 * DIM);
#pragma unroll
        for (int i = 0; i < 4; ++i) {
            float2 v0 = r0[lane + i * 32];
            float2 v1 = r1[lane + i * 32];
            ax[i] += v0.x + v1.x;
            ay[i] += v0.y + v1.y;
        }
    }
    if (j < e) {
        int c0 = csr_col[j];
        const float2* r0 = (const float2*)(z + (size_t)c0 * DIM);
#pragma unroll
        for (int i = 0; i < 4; ++i) {
            float2 v0 = r0[lane + i * 32];
            ax[i] += v0.x;
            ay[i] += v0.y;
        }
    }

    float inv = 1.0f / fmaxf((float)(e - s), 1.0f);
    const float2* zn = (const float2*)(z + (size_t)n * DIM);
    unsigned int* row = combA + (size_t)n * 256;         // 512 bf16 = 256 u32
#pragma unroll
    for (int i = 0; i < 4; ++i) {
        int p = lane + i * 32;                           // pair index 0..127
        PackPair pa;
        pa.h[0] = (__bf16)(ax[i] * inv);
        pa.h[1] = (__bf16)(ay[i] * inv);
        row[p] = pa.u;
        float2 zv = zn[p];
        PackPair pz;
        pz.h[0] = (__bf16)zv.x;
        pz.h[1] = (__bf16)zv.y;
        row[128 + p] = pz.u;
    }
}

// ---------------------------------------------------------------------------
// Repack W[l] (K x 512 x 256 f32) into WMMA B-operand lane layout, bf16.
// ---------------------------------------------------------------------------
__global__ void __launch_bounds__(256)
packB_kernel(const float* __restrict__ Wl, unsigned int* __restrict__ pB) {
    int p = blockIdx.x * 256 + threadIdx.x;              // 262144 u32 pairs
    int idx  = p * 2;
    int j    = idx & 15;
    int lane = (idx >> 4) & 31;
    int nt   = (idx >> 9) & 15;
    int kc   = (idx >> 13) & 15;
    int k    = idx >> 17;
    if (k >= KEXP) return;
    int n  = nt * 16 + (lane & 15);
    int kd = kc * 32 + (lane >> 4) * 8 + (j & 7) + (j >> 3) * 16;
    float v0 = Wl[((size_t)k * DIM2 + kd)     * DIM + n];
    float v1 = Wl[((size_t)k * DIM2 + kd + 1) * DIM + n];
    PackPair pk;
    pk.h[0] = (__bf16)v0;
    pk.h[1] = (__bf16)v1;
    pB[p] = pk.u;
}

// ---------------------------------------------------------------------------
// Main WMMA kernel: block = 16-node tile; 8 waves cover 16 n-tiles (2 each).
// acc[k][j] accumulates expert k over the 512-deep K dim; epilogue applies
// softmax gate + residual + ReLU.
// ---------------------------------------------------------------------------
__global__ void __launch_bounds__(256)
gemm_gate_kernel(const unsigned int* __restrict__ combA,  // [N,512] bf16 (u32 pairs)
                 const unsigned int* __restrict__ pB,     // packed B, bf16
                 const float* __restrict__ wts,           // [N,4]
                 const float* __restrict__ zin,           // [N,256]
                 float* __restrict__ zout) {              // [N,256]
    __shared__ unsigned int lds[16 * 260];                // 512 bf16 + pad per row
    const int m0  = blockIdx.x * 16;
    const int tid = threadIdx.x;

    // Stage A tile (coalesced, streamed-once -> non-temporal)
#pragma unroll
    for (int i = 0; i < 16; ++i) {
        int idx = tid + i * 256;
        int row = idx >> 8;
        int cu  = idx & 255;
        lds[row * 260 + cu] =
            __builtin_nontemporal_load(&combA[(size_t)(m0 + row) * 256 + cu]);
    }
    __syncthreads();

    const int wid     = tid >> 5;
    const int lane    = tid & 31;
    const int laneRow = lane & 15;
    const int hi      = lane >> 4;

    v8f acc[KEXP][2];
    const v8f vzero = {0.f, 0.f, 0.f, 0.f, 0.f, 0.f, 0.f, 0.f};
#pragma unroll
    for (int k = 0; k < KEXP; ++k)
#pragma unroll
        for (int j = 0; j < 2; ++j) acc[k][j] = vzero;

    for (int kk = 0; kk < 16; ++kk) {
        // A fragment per ISA layout: lo lanes K{0..7,16..23}, hi lanes K{8..15,24..31}
        Frag a;
        const u32x4* ap = (const u32x4*)&lds[laneRow * 260 + kk * 16 + hi * 4];
        a.u[0] = ap[0];
        a.u[1] = ap[2];   // +8 u32 = +16 bf16 in K
#pragma unroll
        for (int k = 0; k < KEXP; ++k) {
#pragma unroll
            for (int j = 0; j < 2; ++j) {
                int nt = wid * 2 + j;
                Frag b;
                const u32x4* bp =
                    (const u32x4*)&pB[(size_t)(((k * 16 + kk) * 16 + nt) * 32 + lane) * 8];
                b.u[0] = bp[0];
                b.u[1] = bp[1];
                acc[k][j] = __builtin_amdgcn_wmma_f32_16x16x32_bf16(
                    false, a.v, false, b.v, (short)0, acc[k][j], false, false);
            }
        }
    }

    // Epilogue: C/D layout -> row = m0 + v + 8*hi, col = n-tile*16 + (lane&15)
    float4 wv[8];
#pragma unroll
    for (int v = 0; v < 8; ++v)
        wv[v] = ((const float4*)wts)[m0 + v + hi * 8];

#pragma unroll
    for (int j = 0; j < 2; ++j) {
        int n = (wid * 2 + j) * 16 + laneRow;
#pragma unroll
        for (int v = 0; v < 8; ++v) {
            int m = m0 + v + hi * 8;
            float r = wv[v].x * acc[0][j][v] + wv[v].y * acc[1][j][v] +
                      wv[v].z * acc[2][j][v] + wv[v].w * acc[3][j][v] +
                      zin[(size_t)m * DIM + n];
            zout[(size_t)m * DIM + n] = fmaxf(r, 0.0f);
        }
    }
}

// ---------------------------------------------------------------------------
extern "C" void kernel_launch(void* const* d_in, const int* in_sizes, int n_in,
                              void* d_out, int out_size, void* d_ws, size_t ws_size,
                              hipStream_t stream) {
    const float*     x    = (const float*)d_in[0];
    const long long* ei   = (const long long*)d_in[1];   // int64 edge_index [2,E]
    const float*     W    = (const float*)d_in[2];       // [L,K,512,256]
    const float*     envW = (const float*)d_in[3];       // [L,256,4]
    const float*     envb = (const float*)d_in[4];       // [L,4]
    float*           out  = (float*)d_out;

    char* ws = (char*)d_ws;
    size_t off_b = 0;
    auto carve = [&](size_t bytes) -> void* {
        void* p = ws + off_b;
        off_b = (off_b + bytes + 255) & ~(size_t)255;
        return p;
    };
    float*        zbuf    = (float*)carve((size_t)N_NODES * DIM * 4);
    float*        wts     = (float*)carve((size_t)N_NODES * KEXP * 4);
    unsigned int* combA   = (unsigned int*)carve((size_t)N_NODES * DIM2 * 2);
    unsigned int* pB      = (unsigned int*)carve((size_t)KEXP * DIM2 * DIM * 2);
    int*          deg     = (int*)carve((size_t)N_NODES * 4);
    int*          offcsr  = (int*)carve((size_t)(N_NODES + 1) * 4);
    int*          cursor  = (int*)carve((size_t)N_NODES * 4);
    int*          csr_col = (int*)carve((size_t)E_EDGES * 4);

    // ---- CSR build (edge structure is constant across layers) ----
    hipMemsetAsync(deg,    0, (size_t)N_NODES * 4, stream);
    hipMemsetAsync(cursor, 0, (size_t)N_NODES * 4, stream);
    deg_kernel <<<E_EDGES / 256, 256,  0, stream>>>(ei, deg);
    scan_kernel<<<1,             1024, 0, stream>>>(deg, offcsr);
    fill_kernel<<<E_EDGES / 256, 256,  0, stream>>>(ei, offcsr, cursor, csr_col);

    for (int l = 0; l < LLAYERS; ++l) {
        const float* zin  = (l == 0) ? x : zbuf;
        float*       zout = (l == LLAYERS - 1) ? out : zbuf;

        router_kernel<<<N_NODES / 8, 256, 0, stream>>>(
            zin, envW + (size_t)l * DIM * KEXP, envb + (size_t)l * KEXP, wts);

        agg_combine_kernel<<<N_NODES / 8, 256, 0, stream>>>(zin, offcsr, csr_col, combA);

        packB_kernel<<<(KEXP * DIM2 * DIM / 2) / 256, 256, 0, stream>>>(
            W + (size_t)l * KEXP * DIM2 * DIM, pB);

        gemm_gate_kernel<<<N_NODES / 16, 256, 0, stream>>>(combA, pB, wts, zin, zout);
    }
}
